// VectorQuantizer_78537771974770
// MI455X (gfx1250) — matrix operand order repaired and checked by reading, not compiled, not run on hardware
//
#include <hip/hip_runtime.h>
#include <hip/hip_bf16.h>

#define EMBED_DIM 256
#define NUM_EMBED 8192
#define N_TOKENS  32768
#define NTILES    (NUM_EMBED / 16)

typedef _Float16 v4h  __attribute__((ext_vector_type(4)));
typedef _Float16 v8h  __attribute__((ext_vector_type(8)));
typedef _Float16 v16h __attribute__((ext_vector_type(16)));
typedef float    v8f  __attribute__((ext_vector_type(8)));
typedef int      v4i  __attribute__((ext_vector_type(4)));

union HU { v16h v; v8h h[2]; };

// ---------------------------------------------------------------------------
// CDNA5 async global->LDS copy (ASYNCcnt-tracked) + wait
// ---------------------------------------------------------------------------
__device__ __forceinline__ void async_copy_b128(const void* g, void* l) {
#if __has_builtin(__builtin_amdgcn_global_load_async_to_lds_b128)
    typedef __attribute__((address_space(1))) v4i* gptr_t;   // global (AS1)
    typedef __attribute__((address_space(3))) v4i* lptr_t;   // LDS (AS3)
    __builtin_amdgcn_global_load_async_to_lds_b128((gptr_t)g, (lptr_t)l, 0, 0);
#else
    // VDST = LDS byte address, VADDR = 64-bit global address (GV mode)
    unsigned off = (unsigned)(uintptr_t)l;  // low 32 bits of flat LDS addr
    asm volatile("global_load_async_to_lds_b128 %0, %1, off"
                 :: "v"(off), "v"(g) : "memory");
#endif
}

__device__ __forceinline__ void wait_async0() {
#if __has_builtin(__builtin_amdgcn_s_wait_asynccnt)
    __builtin_amdgcn_s_wait_asynccnt(0);
#else
    asm volatile("s_wait_asynccnt 0x0" ::: "memory");
#endif
}

// ---------------------------------------------------------------------------
// Kernel 1: prep — split codebook row into f16 hi/lo halves (cs[k][0..255]=hi,
// cs[k][256..511]=lo, contiguous 1KB per row) and compute cn[k] = ||e_k||^2.
// ---------------------------------------------------------------------------
__global__ void vq_prep(const float* __restrict__ cb, _Float16* __restrict__ cs,
                        float* __restrict__ cn) {
    __shared__ float red[256];
    const int row = blockIdx.x;
    const int t   = threadIdx.x;
    const float v = cb[(size_t)row * EMBED_DIM + t];
    const _Float16 h = (_Float16)v;
    cs[(size_t)row * 512 + t]       = h;
    cs[(size_t)row * 512 + 256 + t] = (_Float16)(v - (float)h);
    red[t] = v * v;
    __syncthreads();
    for (int s = 128; s > 0; s >>= 1) {
        if (t < s) red[t] += red[t + s];
        __syncthreads();
    }
    if (t == 0) cn[row] = red[0];
}

// ---------------------------------------------------------------------------
// Kernel 2: fused distance GEMM (split-f16 WMMA) + running argmin.
//   Block: 256 threads = 8 waves, 128 tokens. Codebook tiles (16 codes,
//   pre-split hi/lo f16) stream via async global->LDS copies, double-buffered.
// ---------------------------------------------------------------------------
__global__ __launch_bounds__(256) void vq_argmin(const float* __restrict__ x,
                                                 const _Float16* __restrict__ cs,
                                                 const float* __restrict__ cn,
                                                 int* __restrict__ idx_out) {
    // row = [hi 256 | lo 256 | pad 8] halves -> 1040B stride: bank offset 4/row,
    // so the 16 B-fragment lanes hit disjoint bank quads on ds_load_b128.
    __shared__ __align__(16) _Float16 cbuf[2][16][520];

    const int tid  = threadIdx.x;
    const int lane = tid & 31;
    const int wave = tid >> 5;
    const int half = lane >> 4;   // 0: lanes 0-15, 1: lanes 16-31
    const int n16  = lane & 15;

    // --- A fragments: 16 tokens per wave, full D=256 cached in registers ---
    // 16-bit A 16x32 layout: lane half selects K base (0 or 8); elements 0..7
    // hold K=base..base+7, elements 8..15 hold K=base+16..base+23.
    const int row = blockIdx.x * 128 + wave * 16 + n16;
    const float* xr = x + (size_t)row * EMBED_DIM;

    v16h Ahi[8], Alo[8];
#pragma unroll
    for (int kc = 0; kc < 8; ++kc) {
        const int base = kc * 32 + half * 8;
#pragma unroll
        for (int i = 0; i < 8; ++i) {
            float v0 = xr[base + i];
            float v1 = xr[base + 16 + i];
            _Float16 h0 = (_Float16)v0;
            _Float16 h1 = (_Float16)v1;
            Ahi[kc][i]     = h0;  Alo[kc][i]     = (_Float16)(v0 - (float)h0);
            Ahi[kc][i + 8] = h1;  Alo[kc][i + 8] = (_Float16)(v1 - (float)h1);
        }
    }

    float best[8];
    int   bidx[8];
#pragma unroll
    for (int i = 0; i < 8; ++i) { best[i] = 3.4e38f; bidx[i] = 0x7fffffff; }

    // async fill of one 16KB tile: 1024 b128 transfers, 4 per thread
    auto fill_async = [&](int buf, int tile) {
#pragma unroll
        for (int j = 0; j < 4; ++j) {
            const int flat = j * 256 + tid;   // 0..1023 16B segments
            const int r    = flat >> 6;       // 64 segments per code row
            const int seg  = flat & 63;
            const char* g = (const char*)cs +
                            ((size_t)(tile * 16 + r) * 1024) + (size_t)seg * 16;
            void* l = (char*)&cbuf[buf][r][0] + seg * 16;
            async_copy_b128(g, l);
        }
    };

    fill_async(0, 0);
    wait_async0();
    __syncthreads();

    for (int tile = 0; tile < NTILES; ++tile) {
        const int buf = tile & 1;
        if (tile + 1 < NTILES) fill_async(buf ^ 1, tile + 1);

        // ---- 16x16 score tile: 8 k-steps x 3 split products ----
        v8f acc = {};
#pragma unroll
        for (int kc = 0; kc < 8; ++kc) {
            const int kb = kc * 32 + half * 8;  // B: col n16, K=kb..kb+7, kb+16..kb+23
            HU bh, bl;
            bh.h[0] = *(const v8h*)&cbuf[buf][n16][kb];
            bh.h[1] = *(const v8h*)&cbuf[buf][n16][kb + 16];
            bl.h[0] = *(const v8h*)&cbuf[buf][n16][256 + kb];
            bl.h[1] = *(const v8h*)&cbuf[buf][n16][256 + kb + 16];
            acc = __builtin_amdgcn_wmma_f32_16x16x32_f16(false, Ahi[kc], false, bh.v,
                                                         (short)0, acc, false, false);
            acc = __builtin_amdgcn_wmma_f32_16x16x32_f16(false, Ahi[kc], false, bl.v,
                                                         (short)0, acc, false, false);
            acc = __builtin_amdgcn_wmma_f32_16x16x32_f16(false, Alo[kc], false, bh.v,
                                                         (short)0, acc, false, false);
        }

        // ---- running argmin update: score = ||e||^2 - 2 * (x . e) ----
        const int   code_g = tile * 16 + n16;
        const float cnv    = cn[code_g];
#pragma unroll
        for (int i = 0; i < 8; ++i) {
            float s = fmaf(-2.0f, acc[i], cnv);
            if (s < best[i] || (s == best[i] && code_g < bidx[i])) {
                best[i] = s;
                bidx[i] = code_g;
            }
        }

        wait_async0();     // next tile's copies (this wave) have landed
        __syncthreads();   // everyone done reading buf + everyone's copies done
    }

    // ---- reduce over the 16 lanes holding row m (C/D layout: half-waves) ----
#pragma unroll
    for (int i = 0; i < 8; ++i) {
#pragma unroll
        for (int off = 8; off >= 1; off >>= 1) {
            float os = __shfl_xor(best[i], off, 16);
            int   oi = __shfl_xor(bidx[i], off, 16);
            if (os < best[i] || (os == best[i] && oi < bidx[i])) {
                best[i] = os;
                bidx[i] = oi;
            }
        }
    }
    if (n16 == 0) {
        // lane 0 -> rows m=0..7, lane 16 -> rows m=8..15
#pragma unroll
        for (int i = 0; i < 8; ++i)
            idx_out[blockIdx.x * 128 + wave * 16 + half * 8 + i] = bidx[i];
    }
}

// ---------------------------------------------------------------------------
// Kernel 3: zero the loss accumulator
// ---------------------------------------------------------------------------
__global__ void vq_init(float* acc) {
    if (threadIdx.x == 0 && blockIdx.x == 0) acc[0] = 0.0f;
}

// ---------------------------------------------------------------------------
// Kernel 4: gather quantized rows + accumulate squared error
// ---------------------------------------------------------------------------
__global__ void vq_gather(const float* __restrict__ x, const float* __restrict__ cb,
                          const int* __restrict__ idx, float* __restrict__ out,
                          float* __restrict__ acc) {
    __shared__ float red[256];
    const int tok = blockIdx.x;
    const int t   = threadIdx.x;
    const int k   = idx[tok];
    const float q  = cb[(size_t)k * EMBED_DIM + t];
    const float xv = x[(size_t)tok * EMBED_DIM + t];
    out[(size_t)tok * EMBED_DIM + t] = q;  // straight-through output == q
    const float d = xv - q;
    red[t] = d * d;
    __syncthreads();
    for (int s = 128; s > 0; s >>= 1) {
        if (t < s) red[t] += red[t + s];
        __syncthreads();
    }
    if (t == 0) atomicAdd(acc, red[0]);
}

// ---------------------------------------------------------------------------
// Kernel 5: finalize loss = (1.0 + 0.25) * mean((q - x)^2)
// ---------------------------------------------------------------------------
__global__ void vq_fin(const float* __restrict__ acc, float* __restrict__ out) {
    out[(size_t)N_TOKENS * EMBED_DIM] =
        acc[0] * 1.25f / ((float)N_TOKENS * (float)EMBED_DIM);
}

// ---------------------------------------------------------------------------
extern "C" void kernel_launch(void* const* d_in, const int* in_sizes, int n_in,
                              void* d_out, int out_size, void* d_ws, size_t ws_size,
                              hipStream_t stream) {
    const float* x  = (const float*)d_in[0];   // [32768, 256] f32
    const float* cb = (const float*)d_in[1];   // [8192, 256]  f32
    float* out = (float*)d_out;                // [32768*256 + 1] f32

    // workspace: cn (8192 f32) | idx (32768 i32) | acc+pad (16B) | cs (8 MB f16)
    char* ws = (char*)d_ws;
    float*     cn  = (float*)ws;
    int*       idx = (int*)(ws + NUM_EMBED * sizeof(float));
    float*     acc = (float*)(ws + NUM_EMBED * sizeof(float) + N_TOKENS * sizeof(int));
    _Float16*  cs  = (_Float16*)(ws + NUM_EMBED * sizeof(float) +
                                 N_TOKENS * sizeof(int) + 16);

    vq_prep<<<NUM_EMBED, 256, 0, stream>>>(cb, cs, cn);
    vq_init<<<1, 64, 0, stream>>>(acc);
    vq_argmin<<<N_TOKENS / 128, 256, 0, stream>>>(x, cs, cn, idx);
    vq_gather<<<N_TOKENS, 256, 0, stream>>>(x, cb, idx, out, acc);
    vq_fin<<<1, 1, 0, stream>>>(acc, out);
}
